// GCN3Layer_PyG_996432412811
// MI455X (gfx1250) — matrix operand order repaired and checked by reading, not compiled, not run on hardware
//
#include <hip/hip_runtime.h>
#include <hip/hip_bf16.h>
#include <math.h>

typedef __attribute__((ext_vector_type(16))) __bf16 v16bf;
typedef __attribute__((ext_vector_type(2)))  __bf16 v2bf;
typedef __attribute__((ext_vector_type(8)))  float  v8f;

#define NH    128     // hidden/feature width
#define LDSW  68      // padded per-column stride (uints): bank = (4*col+j)%64

// Pair-convert two f32 to packed bf16 using the HW cvt op when the builtin is
// declared, else plain casts (clang still selects v_cvt_* when available).
__device__ __forceinline__ v2bf cvt2_bf16(float a, float b) {
#if __has_builtin(__builtin_amdgcn_cvt_pk_bf16_f32)
  return __builtin_amdgcn_cvt_pk_bf16_f32(a, b);
#else
  v2bf r;
  r[0] = (__bf16)a;
  r[1] = (__bf16)b;
  return r;
#endif
}

__device__ __forceinline__ unsigned pack_bf16(float a, float b) {
  union { v2bf v; unsigned u; } p;
  p.v = cvt2_bf16(a, b);
  return p.u;
}

// ---------------- degree / norm ----------------
__global__ __launch_bounds__(256) void k_fill1(float* p, int n) {
  int i = blockIdx.x * 256 + threadIdx.x;
  if (i < n) p[i] = 1.0f;   // self-loop contributes 1 to every degree
}

__global__ __launch_bounds__(256) void k_deg(const int* __restrict__ dst,
                                             float* deg, int ne) {
  int i = blockIdx.x * 256 + threadIdx.x;
  if (i < ne) atomicAdd(&deg[dst[i]], 1.0f);
}

__global__ __launch_bounds__(256) void k_rsqrt(float* p, int n) {
  int i = blockIdx.x * 256 + threadIdx.x;
  if (i < n) p[i] = rsqrtf(p[i]);   // dis = deg^-1/2
}

// ---------------- GEMM: Z = act(A) @ W, optionally Out = bias + Z*dis^2 -----
// 256 threads = 8 waves; block computes 128 rows; wave computes 16 rows x 128.
// bf16 WMMA 16x16x32, fp32 accumulate. RELU (template) folds the previous
// layer's ReLU into the A load. OOB rows are clamped (branch-free loads; a
// clamped A row only corrupts its own unstored C row). LDS weight layout is
// [col][k2] (stride 68) so each B fragment is two contiguous ds_load_b128.
template <bool RELU>
__global__ __launch_bounds__(256) void k_gemm(const float* __restrict__ A,
                                              const float* __restrict__ W,
                                              float* __restrict__ Z,
                                              int nrows,
                                              const float* __restrict__ dis,
                                              const float* __restrict__ bias,
                                              float* __restrict__ Out) {
  __shared__ unsigned lds_w[NH * LDSW];   // [col][k2] packed bf16x2

  int tid = threadIdx.x;
  // cooperative load + f32->bf16 pack of the 128x128 weight tile (transposed)
  for (int idx = tid; idx < 64 * 128; idx += 256) {
    int k2 = idx >> 7;        // K-pair index 0..63
    int n  = idx & 127;       // output column
    float w0 = W[(2 * k2) * NH + n];
    float w1 = W[(2 * k2 + 1) * NH + n];
    lds_w[n * LDSW + k2] = pack_bf16(w0, w1);
  }
  __syncthreads();

  int lane = tid & 31;
  int wv   = tid >> 5;
  int m    = lane & 15;       // row (A) / col (B,C) within tile
  int hi   = lane >> 4;       // lane-half selects K/M sub-block per ISA layout

  long long rowA = (long long)blockIdx.x * 128 + wv * 16 + m;
  if (rowA >= nrows) rowA = nrows - 1;           // clamp: branch-free loads
  const float* arow = A + rowA * NH;

  v8f acc[8] = {};            // 8 column tiles of 16

#pragma unroll
  for (int kk = 0; kk < NH; kk += 32) {
    if (kk + 32 < NH)
      __builtin_prefetch(arow + kk + 32, 0, 3);  // global_prefetch_b8

    // ---- A fragment: lane holds row rowA.
    // low lanes : VGPR0..3 = K {kk+0..7 pairs},  VGPR4..7 = K {kk+16..23}
    // high lanes: VGPR0..3 = K {kk+8..15},       VGPR4..7 = K {kk+24..31}
    int k0 = kk + hi * 8;
    float4 a0 = *reinterpret_cast<const float4*>(arow + k0);
    float4 a1 = *reinterpret_cast<const float4*>(arow + k0 + 4);
    float4 a2 = *reinterpret_cast<const float4*>(arow + k0 + 16);
    float4 a3 = *reinterpret_cast<const float4*>(arow + k0 + 20);
    if (RELU) {
      a0.x = fmaxf(a0.x, 0.f); a0.y = fmaxf(a0.y, 0.f);
      a0.z = fmaxf(a0.z, 0.f); a0.w = fmaxf(a0.w, 0.f);
      a1.x = fmaxf(a1.x, 0.f); a1.y = fmaxf(a1.y, 0.f);
      a1.z = fmaxf(a1.z, 0.f); a1.w = fmaxf(a1.w, 0.f);
      a2.x = fmaxf(a2.x, 0.f); a2.y = fmaxf(a2.y, 0.f);
      a2.z = fmaxf(a2.z, 0.f); a2.w = fmaxf(a2.w, 0.f);
      a3.x = fmaxf(a3.x, 0.f); a3.y = fmaxf(a3.y, 0.f);
      a3.z = fmaxf(a3.z, 0.f); a3.w = fmaxf(a3.w, 0.f);
    }
    union { v2bf p[8]; v16bf v; } af;
    af.p[0] = cvt2_bf16(a0.x, a0.y);
    af.p[1] = cvt2_bf16(a0.z, a0.w);
    af.p[2] = cvt2_bf16(a1.x, a1.y);
    af.p[3] = cvt2_bf16(a1.z, a1.w);
    af.p[4] = cvt2_bf16(a2.x, a2.y);
    af.p[5] = cvt2_bf16(a2.z, a2.w);
    af.p[6] = cvt2_bf16(a3.x, a3.y);
    af.p[7] = cvt2_bf16(a3.z, a3.w);

    // ---- 8 column tiles: B fragment = two 4-uint runs per lane in LDS.
    // lane col = t*16 + m; low lanes take k2 {kk/2+0..3, +8..11}, high {kk/2+4..7, +12..15}
    int k2base = (kk >> 1) + hi * 4;
#pragma unroll
    for (int t = 0; t < 8; ++t) {
      union { uint4 q[2]; v16bf v; } bf;
      const uint4* bp = reinterpret_cast<const uint4*>(
          &lds_w[(t * 16 + m) * LDSW + k2base]);
      bf.q[0] = bp[0];
      bf.q[1] = bp[2];
      acc[t] = __builtin_amdgcn_wmma_f32_16x16x32_bf16(
          false, af.v, false, bf.v, (short)0, acc[t], false, false);
    }
  }

  // ---- epilogue: C layout: VGPR v -> row (hi*8 + v), col = t*16 + m
  float bv[8];
  if (Out) {
#pragma unroll
    for (int t = 0; t < 8; ++t) bv[t] = bias[t * 16 + m];
  }
  long long rbase = (long long)blockIdx.x * 128 + wv * 16 + hi * 8;
#pragma unroll
  for (int v = 0; v < 8; ++v) {
    long long r = rbase + v;
    if (r < nrows) {
      float* zr = Z + r * NH;
#pragma unroll
      for (int t = 0; t < 8; ++t) zr[t * 16 + m] = acc[t][v];
      if (Out) {                       // fused: Out = bias + self-loop term
        float d  = dis[r];
        float dd = d * d;
        float* orr = Out + r * NH;
#pragma unroll
        for (int t = 0; t < 8; ++t) orr[t * 16 + m] = bv[t] + acc[t][v] * dd;
      }
    }
  }
}

// ---------------- aggregation init (fallback when ws is small) --------------
__global__ __launch_bounds__(256) void k_init_out(const float* __restrict__ Z,
                                                  const float* __restrict__ dis,
                                                  const float* __restrict__ bias,
                                                  float* __restrict__ Out,
                                                  int total) {
  int i = blockIdx.x * 256 + threadIdx.x;
  if (i >= total) return;
  int n = i >> 7;
  int f = i & 127;
  float d = dis[n];
  Out[i] = bias[f] + Z[i] * d * d;   // self-loop norm = 1/deg
}

// ---------------- edge scatter: wave per edge, float4 per lane --------------
__global__ __launch_bounds__(256) void k_scatter(const float* __restrict__ Z,
                                                 const int* __restrict__ src,
                                                 const int* __restrict__ dst,
                                                 const float* __restrict__ dis,
                                                 float* __restrict__ Out, int ne) {
  int e = blockIdx.x * 8 + (threadIdx.x >> 5);
  if (e >= ne) return;
  int lane = threadIdx.x & 31;
  int s = src[e];
  int d = dst[e];
  float nm = dis[s] * dis[d];
  const float4 v =
      *reinterpret_cast<const float4*>(Z + (long long)s * NH + lane * 4);
  float* o = Out + (long long)d * NH + lane * 4;
  atomicAdd(o + 0, v.x * nm);
  atomicAdd(o + 1, v.y * nm);
  atomicAdd(o + 2, v.z * nm);
  atomicAdd(o + 3, v.w * nm);
}

// ---------------- head: logits = H @ Wl + bl, then log_softmax --------------
__global__ __launch_bounds__(256) void k_head(const float* __restrict__ H,
                                              const float* __restrict__ Wl,
                                              const float* __restrict__ bl,
                                              float* __restrict__ Out, int n) {
  int node = blockIdx.x * 8 + (threadIdx.x >> 5);
  if (node >= n) return;
  int lane = threadIdx.x & 31;
  float4 hv =
      *reinterpret_cast<const float4*>(H + (long long)node * NH + lane * 4);
  float acc[10];
#pragma unroll
  for (int c = 0; c < 10; ++c) acc[c] = 0.f;
  int k = lane * 4;
#pragma unroll
  for (int j = 0; j < 4; ++j) {
    float h = (&hv.x)[j];
    const float* wr = Wl + (k + j) * 10;
#pragma unroll
    for (int c = 0; c < 10; ++c) acc[c] += h * wr[c];
  }
#pragma unroll
  for (int off = 16; off >= 1; off >>= 1) {
#pragma unroll
    for (int c = 0; c < 10; ++c) acc[c] += __shfl_xor(acc[c], off, 32);
  }
  if (lane == 0) {
    float mx = -1e30f;
#pragma unroll
    for (int c = 0; c < 10; ++c) {
      acc[c] += bl[c];
      mx = fmaxf(mx, acc[c]);
    }
    float s = 0.f;
#pragma unroll
    for (int c = 0; c < 10; ++c) s += __expf(acc[c] - mx);
    float lse = mx + __logf(s);
    float* o = Out + (long long)node * 10;
#pragma unroll
    for (int c = 0; c < 10; ++c) o[c] = acc[c] - lse;
  }
}

extern "C" void kernel_launch(void* const* d_in, const int* in_sizes, int n_in,
                              void* d_out, int out_size, void* d_ws,
                              size_t ws_size, hipStream_t stream) {
  const float* x  = (const float*)d_in[0];
  const int*   ei = (const int*)d_in[1];
  const float* W1 = (const float*)d_in[2];
  const float* b1 = (const float*)d_in[3];
  const float* W2 = (const float*)d_in[4];
  const float* b2 = (const float*)d_in[5];
  const float* W3 = (const float*)d_in[6];
  const float* b3 = (const float*)d_in[7];
  const float* Wl = (const float*)d_in[8];
  const float* bl = (const float*)d_in[9];
  float* out = (float*)d_out;

  int N = in_sizes[0] / NH;
  int E = in_sizes[1] / 2;
  const int* src = ei;         // edge_index row 0
  const int* dst = ei + E;     // edge_index row 1

  size_t nh = (size_t)N * NH;
  bool fused = ws_size >= (3 * nh + (size_t)N) * sizeof(float);

  float* h0  = (float*)d_ws;                       // N x 128
  float* h1  = h0 + nh;                            // N x 128
  float* h2  = fused ? (h1 + nh) : nullptr;        // N x 128 (fused only)
  float* dis = fused ? (h2 + nh) : (h1 + nh);      // N

  dim3 b256(256);
  int gN    = (N + 255) / 256;
  int gE    = (E + 255) / 256;
  int gGemm = (N + 127) / 128;
  int gElem = (N * NH + 255) / 256;
  int gEdge = (E + 7) / 8;
  int gHead = (N + 7) / 8;

  // normalization: deg (with self loops) -> dis = deg^-1/2
  k_fill1<<<gN, b256, 0, stream>>>(dis, N);
  k_deg<<<gE, b256, 0, stream>>>(dst, dis, E);
  k_rsqrt<<<gN, b256, 0, stream>>>(dis, N);

  if (fused) {
    // GEMM writes Z and the bias+self-loop-initialized aggregation buffer.
    // Buffers rotate so Out never aliases the GEMM input.
    k_gemm<false><<<gGemm, b256, 0, stream>>>(x,  W1, h1, N, dis, b1, h0);
    k_scatter<<<gEdge, b256, 0, stream>>>(h1, src, dst, dis, h0, E);

    k_gemm<true><<<gGemm, b256, 0, stream>>>(h0, W2, h1, N, dis, b2, h2);
    k_scatter<<<gEdge, b256, 0, stream>>>(h1, src, dst, dis, h2, E);

    k_gemm<true><<<gGemm, b256, 0, stream>>>(h2, W3, h1, N, dis, b3, h0);
    k_scatter<<<gEdge, b256, 0, stream>>>(h1, src, dst, dis, h0, E);
  } else {
    // 2-buffer fallback: separate aggregation-init pass.
    k_gemm<false><<<gGemm, b256, 0, stream>>>(x, W1, h1, N, dis, b1, nullptr);
    k_init_out<<<gElem, b256, 0, stream>>>(h1, dis, b1, h0, N * NH);
    k_scatter<<<gEdge, b256, 0, stream>>>(h1, src, dst, dis, h0, E);

    k_gemm<true><<<gGemm, b256, 0, stream>>>(h0, W2, h1, N, dis, b2, nullptr);
    k_init_out<<<gElem, b256, 0, stream>>>(h1, dis, b2, h0, N * NH);
    k_scatter<<<gEdge, b256, 0, stream>>>(h1, src, dst, dis, h0, E);

    k_gemm<true><<<gGemm, b256, 0, stream>>>(h0, W3, h1, N, dis, b3, nullptr);
    k_init_out<<<gElem, b256, 0, stream>>>(h1, dis, b3, h0, N * NH);
    k_scatter<<<gEdge, b256, 0, stream>>>(h1, src, dst, dis, h0, E);
  }

  // head + log_softmax (layer 3 has no relu)
  k_head<<<gHead, b256, 0, stream>>>(h0, Wl, bl, out, N);
}
